// PerTokenFFN_5970004542166
// MI455X (gfx1250) — matrix operand order repaired and checked by reading, not compiled, not run on hardware
//
#include <hip/hip_runtime.h>
#include <math.h>

// Problem constants (match reference)
#define BB 16384
#define TT 128
#define DD 32
#define HH 128
#define TD (TT * DD)   // 4096
#define ROWS_PER_BLOCK 128
#define THREADS 256    // 8 wave32

typedef __bf16 v16bf __attribute__((ext_vector_type(16)));
typedef float  v8f   __attribute__((ext_vector_type(8)));

// Branchless tanh-form GELU using gfx1250 trans ops (v_exp_f32 + v_rcp_f32).
// tanh(y) = 1 - 2/(exp(2y)+1); saturates correctly at +/-inf without branches.
__device__ __forceinline__ float gelu_fast(float x) {
    float y  = 0.7978845608028654f * (x + 0.044715f * x * x * x);
    float e  = __builtin_amdgcn_exp2f(y * 2.8853900817779268f); // 2*log2(e)*y
    float th = 1.0f - 2.0f * __builtin_amdgcn_rcpf(e + 1.0f);
    return 0.5f * x * (1.0f + th);
}

__global__ __launch_bounds__(THREADS)
void PerTokenFFN_kernel(const float* __restrict__ x,
                        const float* __restrict__ W1,
                        const float* __restrict__ b1,
                        const float* __restrict__ W2,
                        const float* __restrict__ b2,
                        float* __restrict__ out) {
    const int t       = blockIdx.y;                  // expert index
    const int rowBase = blockIdx.x * ROWS_PER_BLOCK; // batch tile start
    const int tid     = threadIdx.x;
    const int wave    = tid >> 5;                    // 0..7
    const int lane    = tid & 31;
    const int r       = lane & 15;                   // row (A) / col (B, C/D)
    const int half    = lane >> 4;                   // half-wave selector

    // LDS: weights stored TRANSPOSED so B-operand reads are contiguous per lane.
    __shared__ __bf16 sW1T[HH * DD];                 // [h][d], 8 KB
    __shared__ __bf16 sW2T[DD * HH];                 // [d][h], 8 KB
    __shared__ __bf16 sX  [ROWS_PER_BLOCK * DD];     // [row][d], 8 KB
    __shared__ __bf16 sH  [8][16 * HH];              // per-wave 16x128, 32 KB

    // ---- cooperative stage: fp32 -> bf16 (global reads coalesced) ----
    {
        const float* w1 = W1 + (size_t)t * DD * HH;  // [d][h]
        for (int i = tid; i < DD * HH; i += THREADS) {
            int d = i >> 7, h = i & (HH - 1);        // i = d*H + h
            sW1T[h * DD + d] = (__bf16)w1[i];
        }
        const float* w2 = W2 + (size_t)t * HH * DD;  // [h][d]
        for (int i = tid; i < HH * DD; i += THREADS) {
            int h = i >> 5, d = i & (DD - 1);        // i = h*D + d
            sW2T[d * HH + h] = (__bf16)w2[i];
        }
        for (int i = tid; i < ROWS_PER_BLOCK * DD; i += THREADS) {
            int rr = i >> 5, d = i & (DD - 1);
            sX[i] = (__bf16)x[(size_t)(rowBase + rr) * TD + t * DD + d];
        }
    }
    __syncthreads();

    const int mrow = wave * 16;                      // wave's row strip

    // ---- A operand, GEMM1: lane holds row r; K-groups of 8 contiguous ----
    v16bf a1;
    #pragma unroll
    for (int j = 0; j < 16; ++j) {
        int K = 16 * (j >> 3) + 8 * half + (j & 7);  // contiguous in j&7
        a1[j] = sX[(mrow + r) * DD + K];
    }

    // ---- GEMM1 + gelu: 8 N-blocks of 16 over H ----
    #pragma unroll
    for (int ni = 0; ni < 8; ++ni) {
        // B operand from transposed W1: two K-groups of 8 contiguous bf16 per lane
        v16bf bv;
        #pragma unroll
        for (int j = 0; j < 16; ++j)
            bv[j] = sW1T[(ni * 16 + r) * DD + 16 * (j >> 3) + 8 * half + (j & 7)];

        float bias = b1[t * HH + ni * 16 + r];
        v8f c;
        #pragma unroll
        for (int v = 0; v < 8; ++v) c[v] = bias;

        c = __builtin_amdgcn_wmma_f32_16x16x32_bf16(
                false, a1, false, bv, (short)0, c, false, false);

        // C/D layout: element v -> row (v + 8*half), col (ni*16 + r)
        #pragma unroll
        for (int v = 0; v < 8; ++v)
            sH[wave][(v + 8 * half) * HH + ni * 16 + r] = (__bf16)gelu_fast(c[v]);
    }

    // ---- A operands for GEMM2 (shared across both N-blocks) ----
    v16bf a2v[4];
    #pragma unroll
    for (int kk = 0; kk < 4; ++kk) {
        #pragma unroll
        for (int j = 0; j < 16; ++j) {
            int Ka = kk * 32 + 16 * (j >> 3) + 8 * half + (j & 7);
            a2v[kk][j] = sH[wave][r * HH + Ka];
        }
    }

    // ---- GEMM2: [16 x 128] @ [128 x 32], 2 N-blocks x 4 K-steps ----
    #pragma unroll
    for (int nb = 0; nb < 2; ++nb) {
        float bias2 = b2[t * DD + nb * 16 + r];
        v8f c;
        #pragma unroll
        for (int v = 0; v < 8; ++v) c[v] = bias2;

        #pragma unroll
        for (int kk = 0; kk < 4; ++kk) {
            // B operand from transposed W2: two K-groups of 8 contiguous bf16 per lane
            v16bf b2v;
            #pragma unroll
            for (int j = 0; j < 16; ++j)
                b2v[j] = sW2T[(nb * 16 + r) * HH + kk * 32 + 16 * (j >> 3) + 8 * half + (j & 7)];

            c = __builtin_amdgcn_wmma_f32_16x16x32_bf16(
                    false, a2v[kk], false, b2v, (short)0, c, false, false);
        }

        #pragma unroll
        for (int v = 0; v < 8; ++v)
            out[(size_t)(rowBase + mrow + v + 8 * half) * TD + t * DD + nb * 16 + r] = c[v];
    }
}

extern "C" void kernel_launch(void* const* d_in, const int* in_sizes, int n_in,
                              void* d_out, int out_size, void* d_ws, size_t ws_size,
                              hipStream_t stream) {
    const float* x  = (const float*)d_in[0];
    const float* W1 = (const float*)d_in[1];
    const float* b1 = (const float*)d_in[2];
    const float* W2 = (const float*)d_in[3];
    const float* b2 = (const float*)d_in[4];
    float* out = (float*)d_out;

    dim3 grid(BB / ROWS_PER_BLOCK, TT);  // 128 x 128 blocks
    dim3 block(THREADS);
    PerTokenFFN_kernel<<<grid, block, 0, stream>>>(x, W1, b1, W2, b2, out);
}
